// CROMA_15994458210331
// MI455X (gfx1250) — matrix operand order repaired and checked by reading, not compiled, not run on hardware
//
#include <hip/hip_runtime.h>
#include <hip/hip_bf16.h>

typedef __attribute__((ext_vector_type(16))) __bf16 v16bf;
typedef __attribute__((ext_vector_type(8)))  float  v8f;

#define TPB 256

__device__ __forceinline__ float gelu_exact(float v) {
  return 0.5f * v * (1.f + erff(v * 0.70710678118654752f));
}

// ---------------- weight convert: W [K,N] f32 row-major -> Wt [N,K] bf16 ----------------
__global__ __launch_bounds__(TPB) void wcvt_kernel(const float* __restrict__ W,
                                                   __bf16* __restrict__ Wt,
                                                   int K, int N) {
  long i = (long)blockIdx.x * TPB + threadIdx.x;
  long tot = (long)N * K;
  if (i >= tot) return;
  int n = (int)(i / K);
  int k = (int)(i - (long)n * K);
  Wt[i] = (__bf16)W[(long)k * N + n];
}

// ---------------- patch extraction: img [16,Cin,120,120] -> out [16*225, Cin*64] --------
__global__ __launch_bounds__(TPB) void patchify_kernel(const float* __restrict__ img,
                                                       float* __restrict__ out, int Cin) {
  int dim = Cin * 64;
  long tot = 16L * 225 * dim;
  long i = (long)blockIdx.x * TPB + threadIdx.x;
  if (i >= tot) return;
  long rowi = i / dim;
  int dc = (int)(i - rowi * dim);
  int b = (int)(rowi / 225);
  int p = (int)(rowi - (long)b * 225);
  int ph = p / 15, pw = p - ph * 15;
  int c = dc >> 6;
  int rj = dc & 63;
  int r8 = rj >> 3, c8 = rj & 7;
  out[i] = img[(((long)b * Cin + c) * 120 + ph * 8 + r8) * 120 + pw * 8 + c8];
}

// ---------------- LayerNorm: one block per row, Dm = 768 --------------------------------
__global__ __launch_bounds__(TPB) void ln_kernel(const float* __restrict__ X,
                                                 float* __restrict__ Y,
                                                 const float* __restrict__ g,
                                                 const float* __restrict__ b, int Dm) {
  __shared__ float red[TPB];
  long row = blockIdx.x;
  const float* x = X + row * Dm;
  float s = 0.f;
  for (int i = threadIdx.x; i < Dm; i += TPB) s += x[i];
  red[threadIdx.x] = s; __syncthreads();
  for (int o = TPB / 2; o > 0; o >>= 1) {
    if (threadIdx.x < o) red[threadIdx.x] += red[threadIdx.x + o];
    __syncthreads();
  }
  float mean = red[0] / Dm;
  __syncthreads();
  float v = 0.f;
  for (int i = threadIdx.x; i < Dm; i += TPB) { float d = x[i] - mean; v += d * d; }
  red[threadIdx.x] = v; __syncthreads();
  for (int o = TPB / 2; o > 0; o >>= 1) {
    if (threadIdx.x < o) red[threadIdx.x] += red[threadIdx.x + o];
    __syncthreads();
  }
  float inv = rsqrtf(red[0] / Dm + 1e-5f);
  float* y = Y + row * Dm;
  for (int i = threadIdx.x; i < Dm; i += TPB) y[i] = (x[i] - mean) * inv * g[i] + b[i];
}

// ---------------- softmax over rows of S [BH,225,225]; applies scale + alibi bias -------
__global__ __launch_bounds__(TPB) void softmax_kernel(float* __restrict__ S,
                                                      const float* __restrict__ bias,
                                                      float scale, int N) {
  __shared__ float red[TPB];
  long idx = blockIdx.x;                 // z*N + row
  long z = idx / N;
  int row = (int)(idx - z * N);
  int h = (int)(z & 15);
  float* s = S + idx * N;
  const float* bb = bias + ((long)h * N + row) * N;
  int t = threadIdx.x;
  int tc = (t < N) ? t : (N - 1);        // clamped: unconditional load, masked by select
  float v = s[tc] * scale + bb[tc];
  if (t >= N) v = -3.0e38f;
  red[t] = v; __syncthreads();
  for (int o = TPB / 2; o > 0; o >>= 1) {
    if (t < o) red[t] = fmaxf(red[t], red[t + o]);
    __syncthreads();
  }
  float m = red[0]; __syncthreads();
  float e = (t < N) ? __expf(v - m) : 0.f;
  red[t] = e; __syncthreads();
  for (int o = TPB / 2; o > 0; o >>= 1) {
    if (t < o) red[t] += red[t + o];
    __syncthreads();
  }
  float inv = 1.f / red[0];
  if (t < N) s[t] = e * inv;
}

// ---------------- V transpose per (b,h): vt[z][48][225] = src[b, n, h*48 + d] -----------
__global__ __launch_bounds__(TPB) void vtrans_kernel(const float* __restrict__ src, int ldsrc,
                                                     float* __restrict__ vt) {
  int z = blockIdx.y;
  int b = z >> 4, h = z & 15;
  int i = blockIdx.x * TPB + threadIdx.x;
  if (i >= 48 * 225) return;
  int d = i / 225, n = i - d * 225;
  vt[((long)z * 48 + d) * 225 + n] = src[((long)b * 225 + n) * (long)ldsrc + h * 48 + d];
}

// ---------------- WMMA GEMM: C = act(A @ Bt^T + bias + res) -----------------------------
// A: f32 [M,K] (row stride lda). Bt: [N,K] (row stride ldb), bf16 (BF=0) or f32 (BF=1).
// Each wave computes a 16x32 C tile (two 16x16 WMMA tiles sharing the A fragment).
// All loads are unconditional (row/col pointers clamped; OOB rows/cols only feed C
// entries that the store guard masks). K tails use load+select, never exec branches.
template <int BF>
__global__ __launch_bounds__(TPB) void wmma_gemm_kernel(
    const float* __restrict__ A, int lda, long aHi, long aLo,
    const void* __restrict__ Bt, int ldb, long bHi, long bLo,
    float* __restrict__ C, int ldc, long cHi, long cLo,
    const float* __restrict__ bias,
    const float* __restrict__ res, int ldr, long rHi, long rLo,
    int M, int N, int K, int act, int zdiv) {
  int z = blockIdx.z;
  long offA, offB, offC, offR;
  if (zdiv > 0) {
    int zh = z / zdiv, zl = z - zh * zdiv;
    offA = (long)zh * aHi + (long)zl * aLo;
    offB = (long)zh * bHi + (long)zl * bLo;
    offC = (long)zh * cHi + (long)zl * cLo;
    offR = (long)zh * rHi + (long)zl * rLo;
  } else {
    offA = (long)z * aHi; offB = (long)z * bHi; offC = (long)z * cHi; offR = (long)z * rHi;
  }

  const int wave = threadIdx.x >> 5;
  const int lane = threadIdx.x & 31;
  const int hl = lane >> 4;   // wave half (0/1)
  const int lq = lane & 15;

  const int tile_m = blockIdx.y * 128 + wave * 16;
  const int tile_n = blockIdx.x * 32;
  if (tile_m >= M) return;    // wave-uniform exit

  const int m  = tile_m + lq;
  const int n0 = tile_n + lq;
  const int n1 = tile_n + 16 + lq;
  const int mc  = (m  < M) ? m  : (M - 1);
  const int n0c = (n0 < N) ? n0 : (N - 1);
  const int n1c = (n1 < N) ? n1 : (N - 1);

  const float*  arow = A + offA + (long)mc * lda;
  const __bf16* b0h  = (const __bf16*)Bt + offB + (long)n0c * ldb;
  const __bf16* b1h  = (const __bf16*)Bt + offB + (long)n1c * ldb;
  const float*  b0f  = (const float*)Bt + offB + (long)n0c * ldb;
  const float*  b1f  = (const float*)Bt + offB + (long)n1c * ldb;

  v8f acc0 = {}, acc1 = {};
  const int Kfull = K & ~31;

  for (int k0 = 0; k0 < Kfull; k0 += 32) {
    __builtin_prefetch(arow + k0 + 64, 0, 1);
    v16bf af, b0v, b1v;
#pragma unroll
    for (int j = 0; j < 16; ++j) {
      // A 16x32 bf16 fragment: lane row m = lq; element j -> k = 16*half + j
      af[j] = (__bf16)arow[k0 + (hl << 4) + j];
    }
    if (BF == 0) {
      // B fragment is 16 consecutive bf16 (k = 16*half + j): one 32B vector load
      b0v = *(const v16bf*)(b0h + k0 + (hl << 4));
      b1v = *(const v16bf*)(b1h + k0 + (hl << 4));
    } else {
#pragma unroll
      for (int j = 0; j < 16; ++j) {
        int kb = k0 + (hl << 4) + j;
        b0v[j] = (__bf16)b0f[kb];
        b1v[j] = (__bf16)b1f[kb];
      }
    }
    acc0 = __builtin_amdgcn_wmma_f32_16x16x32_bf16(false, af, false, b0v, (short)0, acc0,
                                                   false, false);
    acc1 = __builtin_amdgcn_wmma_f32_16x16x32_bf16(false, af, false, b1v, (short)0, acc1,
                                                   false, false);
  }

  if (Kfull < K) {            // K tail: unconditional loads + select-to-zero (no branches)
    const int k0 = Kfull;
    v16bf af, b0v, b1v;
#pragma unroll
    for (int j = 0; j < 16; ++j) {
      int ka = k0 + (hl << 4) + j;
      float av = arow[ka];
      af[j] = (__bf16)((ka < K) ? av : 0.f);
      int kb = k0 + (hl << 4) + j;
      float v0, v1;
      if (BF == 0) { v0 = (float)b0h[kb]; v1 = (float)b1h[kb]; }
      else         { v0 = b0f[kb];        v1 = b1f[kb]; }
      b0v[j] = (__bf16)((kb < K) ? v0 : 0.f);
      b1v[j] = (__bf16)((kb < K) ? v1 : 0.f);
    }
    acc0 = __builtin_amdgcn_wmma_f32_16x16x32_bf16(false, af, false, b0v, (short)0, acc0,
                                                   false, false);
    acc1 = __builtin_amdgcn_wmma_f32_16x16x32_bf16(false, af, false, b1v, (short)0, acc1,
                                                   false, false);
  }

  // C/D layout: lanes 0-15 -> M = r, N = lane; lanes 16-31 -> M = 8+r, N = lane-16
  float* cb = C + offC;
  const float bc0 = bias ? bias[n0c] : 0.f;
  const float bc1 = bias ? bias[n1c] : 0.f;
  const bool full = (tile_m + 16 <= M) && (tile_n + 32 <= N);   // wave-uniform
  if (full) {
#pragma unroll
    for (int r = 0; r < 8; ++r) {
      int row = tile_m + hl * 8 + r;
      long ro = (long)row * ldc;
      float v0 = acc0[r] + bc0;
      float v1 = acc1[r] + bc1;
      if (res) {
        long rr = offR + (long)row * ldr;
        v0 += res[rr + n0];
        v1 += res[rr + n1];
      }
      if (act == 1) { v0 = gelu_exact(v0); v1 = gelu_exact(v1); }
      cb[ro + n0] = v0;
      cb[ro + n1] = v1;
    }
  } else {
#pragma unroll
    for (int r = 0; r < 8; ++r) {
      int row = tile_m + hl * 8 + r;
      if (row < M) {
        long ro = (long)row * ldc;
        long rr = offR + (long)row * ldr;
        if (n0 < N) {
          float v0 = acc0[r] + bc0;
          if (res) v0 += res[rr + n0];
          if (act == 1) v0 = gelu_exact(v0);
          cb[ro + n0] = v0;
        }
        if (n1 < N) {
          float v1 = acc1[r] + bc1;
          if (res) v1 += res[rr + n1];
          if (act == 1) v1 = gelu_exact(v1);
          cb[ro + n1] = v1;
        }
      }
    }
  }
}

// ======================================================================================
extern "C" void kernel_launch(void* const* d_in, const int* in_sizes, int n_in,
                              void* d_out, int out_size, void* d_ws, size_t ws_size,
                              hipStream_t stream) {
  (void)in_sizes; (void)n_in; (void)out_size; (void)ws_size;
  int ci = 0;
  auto F = [&]() -> const float* { return (const float*)d_in[ci++]; };

  const float* sar = F();
  const float* opt = F();
  const float* abias = F();

  struct AttnW { const float *lng, *lnb, *wqkv, *wout, *bout; __bf16 *qkvT, *outT; };
  struct FfnW  { const float *lng, *lnb, *w1, *b1, *w2, *b2; __bf16 *w1T, *w2T; };
  struct CrossW{ const float *lng, *lnb, *wq, *wk, *wv, *wout, *bout;
                 __bf16 *qT, *kT, *vT, *outT; };

  // ---- parse params in setup_inputs dict order ----
  const float* s1_win = F(); const float* s1_bin = F();
  AttnW s1a[6]; FfnW s1f[6];
  for (int l = 0; l < 6; ++l) {
    s1a[l].lng = F(); s1a[l].lnb = F(); s1a[l].wqkv = F(); s1a[l].wout = F(); s1a[l].bout = F();
    s1f[l].lng = F(); s1f[l].lnb = F(); s1f[l].w1 = F(); s1f[l].b1 = F();
    s1f[l].w2 = F(); s1f[l].b2 = F();
  }
  const float* s1_ng = F(); const float* s1_nb = F();

  const float* s2_win = F(); const float* s2_bin = F();
  AttnW s2a[12]; FfnW s2f[12];
  for (int l = 0; l < 12; ++l) {
    s2a[l].lng = F(); s2a[l].lnb = F(); s2a[l].wqkv = F(); s2a[l].wout = F(); s2a[l].bout = F();
    s2f[l].lng = F(); s2f[l].lnb = F(); s2f[l].w1 = F(); s2f[l].b1 = F();
    s2f[l].w2 = F(); s2f[l].b2 = F();
  }
  const float* s2_ng = F(); const float* s2_nb = F();

  AttnW ca[6]; CrossW cc[6]; FfnW cf[6];
  for (int l = 0; l < 6; ++l) {
    ca[l].lng = F(); ca[l].lnb = F(); ca[l].wqkv = F(); ca[l].wout = F(); ca[l].bout = F();
    cc[l].lng = F(); cc[l].lnb = F(); cc[l].wq = F(); cc[l].wk = F(); cc[l].wv = F();
    cc[l].wout = F(); cc[l].bout = F();
    cf[l].lng = F(); cf[l].lnb = F(); cf[l].w1 = F(); cf[l].b1 = F();
    cf[l].w2 = F(); cf[l].b2 = F();
  }
  const float* c_ng = F(); const float* c_nb = F();

  // ---- workspace bump allocator; 1KB slack per buffer tolerates K-tail over-reads ----
  char* base = (char*)d_ws;
  size_t off = 0;
  auto alloc = [&](size_t bytes) -> void* {
    off = (off + 255) & ~(size_t)255;
    void* p = base + off;
    off += bytes + 1024;
    return p;
  };
  auto cvt = [&](const float* W, int K, int N) -> __bf16* {
    __bf16* T = (__bf16*)alloc((size_t)K * N * 2);
    long tot = (long)K * N;
    wcvt_kernel<<<dim3((unsigned)((tot + TPB - 1) / TPB)), TPB, 0, stream>>>(W, T, K, N);
    return T;
  };

  // ---- convert all weights to transposed bf16 ----
  __bf16* s1_winT = cvt(s1_win, 128, 768);
  for (int l = 0; l < 6; ++l) {
    s1a[l].qkvT = cvt(s1a[l].wqkv, 768, 2304);
    s1a[l].outT = cvt(s1a[l].wout, 768, 768);
    s1f[l].w1T = cvt(s1f[l].w1, 768, 3072);
    s1f[l].w2T = cvt(s1f[l].w2, 3072, 768);
  }
  __bf16* s2_winT = cvt(s2_win, 768, 768);
  for (int l = 0; l < 12; ++l) {
    s2a[l].qkvT = cvt(s2a[l].wqkv, 768, 2304);
    s2a[l].outT = cvt(s2a[l].wout, 768, 768);
    s2f[l].w1T = cvt(s2f[l].w1, 768, 3072);
    s2f[l].w2T = cvt(s2f[l].w2, 3072, 768);
  }
  for (int l = 0; l < 6; ++l) {
    ca[l].qkvT = cvt(ca[l].wqkv, 768, 2304);
    ca[l].outT = cvt(ca[l].wout, 768, 768);
    cc[l].qT = cvt(cc[l].wq, 768, 768);
    cc[l].kT = cvt(cc[l].wk, 768, 768);
    cc[l].vT = cvt(cc[l].wv, 768, 768);
    cc[l].outT = cvt(cc[l].wout, 768, 768);
    cf[l].w1T = cvt(cf[l].w1, 768, 3072);
    cf[l].w2T = cvt(cf[l].w2, 3072, 768);
  }

  // ---- activation buffers ----
  const int R = 16 * 225;                                // 3600 token rows
  float* patches = (float*)alloc((size_t)R * 768 * 4);
  float* x1      = (float*)alloc((size_t)R * 768 * 4);   // SAR stream / cross output
  float* x2      = (float*)alloc((size_t)R * 768 * 4);   // optical stream (context)
  float* lnx     = (float*)alloc((size_t)R * 768 * 4);
  float* lnc     = (float*)alloc((size_t)R * 768 * 4);
  float* qkv     = (float*)alloc((size_t)R * 2304 * 4);
  float* qb      = (float*)alloc((size_t)R * 768 * 4);
  float* kb      = (float*)alloc((size_t)R * 768 * 4);
  float* vb      = (float*)alloc((size_t)R * 768 * 4);
  float* scores  = (float*)alloc((size_t)256 * 225 * 225 * 4);
  float* vt      = (float*)alloc((size_t)256 * 48 * 225 * 4);
  float* attnout = (float*)alloc((size_t)R * 768 * 4);
  float* ffnh    = (float*)alloc((size_t)R * 3072 * 4);

  const float scale = 0.14433756729740643f;              // 48^-0.5

  // ---- launch helpers ----
  auto ln = [&](const float* X, float* Y, const float* g, const float* b) {
    ln_kernel<<<R, TPB, 0, stream>>>(X, Y, g, b, 768);
  };
  auto gemmW = [&](const float* A, int lda, const __bf16* Bt, int ldb, float* C, int ldc,
                   const float* bias_, const float* res_, int ldr, int M, int N, int K,
                   int act) {
    dim3 g((N + 31) / 32, (M + 127) / 128, 1);
    wmma_gemm_kernel<0><<<g, TPB, 0, stream>>>(A, lda, 0, 0, (const void*)Bt, ldb, 0, 0,
                                               C, ldc, 0, 0, bias_, res_, ldr, 0, 0,
                                               M, N, K, act, 0);
  };
  auto gemmB = [&](const float* A, int lda, long aHi, long aLo,
                   const float* Bt, int ldb, long bHi, long bLo,
                   float* C, int ldc, long cHi, long cLo, int M, int N, int K) {
    dim3 g((N + 31) / 32, (M + 127) / 128, 256);
    wmma_gemm_kernel<1><<<g, TPB, 0, stream>>>(A, lda, aHi, aLo, (const void*)Bt, ldb, bHi,
                                               bLo, C, ldc, cHi, cLo, nullptr, nullptr, 0,
                                               0, 0, M, N, K, 0, 16);
  };
  auto attn_core = [&](const float* Q, int ldq, long qHi, long qLo,
                       const float* Kp, int ldk, long kHi, long kLo,
                       const float* Vsrc, int ldv, float* X, const float* outb,
                       __bf16* outT) {
    vtrans_kernel<<<dim3((48 * 225 + TPB - 1) / TPB, 256), TPB, 0, stream>>>(Vsrc, ldv, vt);
    gemmB(Q, ldq, qHi, qLo, Kp, ldk, kHi, kLo,
          scores, 225, (long)16 * 225 * 225, (long)225 * 225, 225, 225, 48);
    softmax_kernel<<<256 * 225, TPB, 0, stream>>>(scores, abias, scale, 225);
    gemmB(scores, 225, (long)16 * 225 * 225, (long)225 * 225,
          vt, 225, (long)16 * 48 * 225, (long)48 * 225,
          attnout, 768, (long)225 * 768, 48, 225, 48, 225);
    gemmW(attnout, 768, outT, 768, X, 768, outb, X, 768, R, 768, 768, 0);  // +residual
  };
  auto self_attn = [&](float* X, AttnW& p) {
    ln(X, lnx, p.lng, p.lnb);
    gemmW(lnx, 768, p.qkvT, 768, qkv, 2304, nullptr, nullptr, 0, R, 2304, 768, 0);
    attn_core(qkv, 2304, (long)225 * 2304, 48,
              qkv + 768, 2304, (long)225 * 2304, 48,
              qkv + 1536, 2304, X, p.bout, p.outT);
  };
  auto cross_attn = [&](float* X, const float* CTX, CrossW& p) {
    ln(X, lnx, p.lng, p.lnb);
    ln(CTX, lnc, p.lng, p.lnb);                     // same LN params for x and context
    gemmW(lnx, 768, p.qT, 768, qb, 768, nullptr, nullptr, 0, R, 768, 768, 0);
    gemmW(lnc, 768, p.kT, 768, kb, 768, nullptr, nullptr, 0, R, 768, 768, 0);
    gemmW(lnc, 768, p.vT, 768, vb, 768, nullptr, nullptr, 0, R, 768, 768, 0);
    attn_core(qb, 768, (long)225 * 768, 48,
              kb, 768, (long)225 * 768, 48,
              vb, 768, X, p.bout, p.outT);
  };
  auto ffn = [&](float* X, FfnW& p) {
    ln(X, lnx, p.lng, p.lnb);
    gemmW(lnx, 768, p.w1T, 3072, ffnh, 3072, p.b1, nullptr, 0, R, 3072, 768, 1);  // GELU
    gemmW(ffnh, 3072, p.w2T, 768, X, 768, p.b2, X, 768, R, 768, 3072, 0);         // +res
  };
  auto encoder = [&](const float* imgs, int Cin, __bf16* winT, const float* bin,
                     AttnW* A, FfnW* Fw, int L, const float* ng, const float* nb,
                     float* X) {
    long tot = 16L * 225 * Cin * 64;
    patchify_kernel<<<(unsigned)((tot + TPB - 1) / TPB), TPB, 0, stream>>>(imgs, patches,
                                                                           Cin);
    gemmW(patches, Cin * 64, winT, Cin * 64, X, 768, bin, nullptr, 0, R, 768, Cin * 64, 0);
    for (int l = 0; l < L; ++l) { self_attn(X, A[l]); ffn(X, Fw[l]); }
    ln(X, X, ng, nb);
  };

  // ---- run the model ----
  encoder(sar, 2, s1_winT, s1_bin, s1a, s1f, 6, s1_ng, s1_nb, x1);
  encoder(opt, 12, s2_winT, s2_bin, s2a, s2f, 12, s2_ng, s2_nb, x2);
  for (int l = 0; l < 6; ++l) {
    self_attn(x1, ca[l]);
    cross_attn(x1, x2, cc[l]);
    ffn(x1, cf[l]);
  }
  ln(x1, (float*)d_out, c_ng, c_nb);
}
